// GptOSSTransformer_28638841929807
// MI455X (gfx1250) — matrix operand order repaired and checked by reading, not compiled, not run on hardware
//
#include <hip/hip_runtime.h>
#include <hip/hip_bf16.h>
#include <math.h>

// ---------------- model constants ----------------
#define DIMC     2880
#define NHEADS   64
#define NKV      8
#define HDIM     64
#define NEXP     4
#define INTERC   2880
#define SEQC     1024
#define QKVD     5120          // 64*64 + 2*8*64
#define QDIM     4096          // 64*64
#define SM_SCALE 0.125f        // 1/sqrt(64)

// ---------------- WMMA plumbing ----------------
typedef __attribute__((ext_vector_type(16))) __bf16        v16bf;
typedef __attribute__((ext_vector_type(2)))  __bf16        v2bf;
typedef __attribute__((ext_vector_type(8)))  float         v8f;
typedef __attribute__((ext_vector_type(2)))  float         v2f;
typedef __attribute__((ext_vector_type(8)))  unsigned int  v8u;

__device__ __forceinline__ v8f zf8() {
    v8f z;
    #pragma unroll
    for (int i = 0; i < 8; ++i) z[i] = 0.0f;
    return z;
}

__device__ __forceinline__ v8f wmma_bf16(v8u a, v8u b, v8f c) {
    return __builtin_amdgcn_wmma_f32_16x16x32_bf16(
        false, __builtin_bit_cast(v16bf, a),
        false, __builtin_bit_cast(v16bf, b),
        (short)0, c, false, false);
}

// packed f32x2 -> bf16x2 (expect v_cvt_pk_bf16_f32)
__device__ __forceinline__ unsigned int pack2(v2f f) {
    return __builtin_bit_cast(unsigned int, __builtin_convertvector(f, v2bf));
}
__device__ __forceinline__ unsigned int pack_bf(float lo, float hi) {
    v2f f; f[0] = lo; f[1] = hi;
    return pack2(f);
}
__device__ __forceinline__ unsigned short f2bf(float f) {
    unsigned int u = __builtin_bit_cast(unsigned int, f);
    u += 0x7FFFu + ((u >> 16) & 1u);            // round-to-nearest-even
    return (unsigned short)(u >> 16);
}

// ---------------- RMSNorm ----------------
__global__ __launch_bounds__(256)
void rmsnorm_k(const float* __restrict__ x, const float* __restrict__ w,
               float* __restrict__ t) {
    int row = blockIdx.x;
    const float* xr = x + row * DIMC;
    float* tr = t + row * DIMC;
    __shared__ float red[256];
    float ss = 0.f;
    for (int i = threadIdx.x; i < DIMC; i += 256) { float v = xr[i]; ss += v * v; }
    red[threadIdx.x] = ss; __syncthreads();
    for (int s = 128; s > 0; s >>= 1) {
        if (threadIdx.x < s) red[threadIdx.x] += red[threadIdx.x + s];
        __syncthreads();
    }
    float inv = rsqrtf(red[0] / (float)DIMC + 1e-5f);
    for (int i = threadIdx.x; i < DIMC; i += 256) tr[i] = xr[i] * inv * w[i];
}

// ---------------- Generic bf16-WMMA GEMM ----------------
// C[M,N] (+)= rowscale[m] * (A[M,K] @ B[K,N] + bias[n])
// fp32 row-major operands, bf16 WMMA compute.
// 128x64 block tile, BK=32, double-buffered LDS, register-staged prefetch.
// B staged TRANSPOSED ([n][k]) so A and B fragments are contiguous dword
// pairs (ds_load_2addr_b64). Staging is pair-packed: v_cvt_pk_bf16_f32 +
// ds_store_b32, A global loads are coalesced b64.
#define BM 128
#define BN 64
#define BK 32
#define LDA_S  36   // u16 row stride (72B -> 18-bank stride, conflict-free)
#define LDBT_S 36

__global__ __launch_bounds__(256)
void gemm_k(const float* __restrict__ A, const float* __restrict__ B,
            const float* __restrict__ bias, const float* __restrict__ rowscale,
            int rs_stride, float* __restrict__ C,
            int M, int N, int K, int lda, int ldb, int ldc, int accumulate) {
    __shared__ unsigned short sA[2][BM * LDA_S];
    __shared__ unsigned short sBT[2][BN * LDBT_S];
    const int tid  = threadIdx.x;
    const int lane = tid & 31;
    const int wid  = tid >> 5;
    const int waveM = wid & 3;            // 4 stripes of 32 rows
    const int waveN = wid >> 2;           // 2 stripes of 32 cols
    const int bm0 = blockIdx.x * BM;
    const int bn0 = blockIdx.y * BN;
    const int half = lane >> 4;
    const int mn   = lane & 15;

    // staging coordinates: A -> thread owns (row ar0+16j, cols ac2..ac2+1)
    //                      B -> thread owns (k pair 2*(bkp*4+j), col bn)
    const int ar0 = tid >> 4;            // 0..15
    const int ac2 = (tid & 15) * 2;      // 0,2,..,30
    const int bn  = tid & 63;            // 0..63
    const int bkp = tid >> 6;            // 0..3
    const float* ApX = A + (bm0 + ar0) * lda + ac2;  // + 16j*lda + k0
    const float* BpX = B + bn0 + bn;                 // + (k0+k)*ldb

    v8f acc[2][2];
    #pragma unroll
    for (int i = 0; i < 2; ++i)
        #pragma unroll
        for (int j = 0; j < 2; ++j) acc[i][j] = zf8();

    v2f ra2[8], rb2[4];

    auto load_tiles = [&](int k0) {
        #pragma unroll
        for (int j = 0; j < 8; ++j)
            ra2[j] = *(const v2f*)&ApX[(16 * j) * lda + k0];
        #pragma unroll
        for (int j = 0; j < 4; ++j) {
            int k = 2 * (bkp * 4 + j);
            v2f r;
            r[0] = BpX[(k0 + k) * ldb];
            r[1] = BpX[(k0 + k + 1) * ldb];
            rb2[j] = r;
        }
    };
    auto store_tiles = [&](int b) {
        unsigned int* a32 = (unsigned int*)&sA[b][0];
        unsigned int* bt32 = (unsigned int*)&sBT[b][0];
        #pragma unroll
        for (int j = 0; j < 8; ++j)
            a32[(ar0 + 16 * j) * (LDA_S / 2) + (ac2 >> 1)] = pack2(ra2[j]);
        #pragma unroll
        for (int j = 0; j < 4; ++j) {
            int k = 2 * (bkp * 4 + j);
            bt32[bn * (LDBT_S / 2) + (k >> 1)] = pack2(rb2[j]);
        }
    };

    // prologue: stage tile k0=0
    load_tiles(0);
    store_tiles(0);
    __syncthreads();

    int buf = 0;
    for (int k0 = 0; k0 < K; k0 += BK) {
        const bool more = (k0 + BK) < K;
        if (more) load_tiles(k0 + BK);   // latency hidden behind WMMAs

        const unsigned short* a0 = &sA[buf][(waveM * 32 + mn) * LDA_S];
        const unsigned short* b0 = &sBT[buf][(waveN * 32 + mn) * LDBT_S];
        v8u afr[2], bfr[2];
        #pragma unroll
        for (int s = 0; s < 2; ++s) {
            v8u a, b;
            #pragma unroll
            for (int v = 0; v < 8; ++v) {
                int k = 2 * (v & 3) + 16 * (v >> 2) + 8 * half;
                a[v] = *(const unsigned int*)&a0[s * 16 * LDA_S + k];
            }
            afr[s] = a;
            #pragma unroll
            for (int v = 0; v < 8; ++v) {
                int k = 2 * (v & 3) + 16 * (v >> 2) + 8 * half;
                b[v] = *(const unsigned int*)&b0[s * 16 * LDBT_S + k];
            }
            bfr[s] = b;
        }
        #pragma unroll
        for (int i = 0; i < 2; ++i)
            #pragma unroll
            for (int j = 0; j < 2; ++j)
                acc[i][j] = wmma_bf16(afr[i], bfr[j], acc[i][j]);

        if (more) store_tiles(buf ^ 1);
        __syncthreads();
        buf ^= 1;
    }

    #pragma unroll
    for (int i = 0; i < 2; ++i)
        #pragma unroll
        for (int j = 0; j < 2; ++j) {
            const int col = bn0 + waveN * 32 + j * 16 + mn;
            const float bv = bias ? bias[col] : 0.f;
            #pragma unroll
            for (int v = 0; v < 8; ++v) {
                const int row = bm0 + waveM * 32 + i * 16 + v + 8 * half;
                float val = acc[i][j][v] + bv;
                if (rowscale) val *= rowscale[row * rs_stride];
                const int off = row * ldc + col;
                if (accumulate) val += C[off];
                C[off] = val;
            }
        }
}

// ---------------- YaRN RoPE (tables computed analytically) ----------------
__global__ __launch_bounds__(32)
void rope_k(float* __restrict__ qkv) {
    const int pos = blockIdx.x;
    const int hh  = blockIdx.y;     // 0..63 q heads, 64..71 kv heads
    const int i   = threadIdx.x;    // 0..31
    const int off = pos * QKVD + (hh < 64 ? hh * 64 : QDIM + (hh - 64) * 64);
    const float lbase = logf(150000.f);
    const float freq  = __expf(((float)(2 * i) / 64.f) * lbase);
    const float TWO_PI = 6.283185307179586f;
    const float lw = 32.f * logf(4096.f / (32.f * TWO_PI)) / lbase;
    const float hi = 32.f * logf(4096.f / (1.f  * TWO_PI)) / lbase;
    float ramp = ((float)i - lw) / (hi - lw);
    float m = 1.f - fminf(fmaxf(ramp, 0.f), 1.f);
    float inv = (1.f / (32.f * freq)) * (1.f - m) + (1.f / freq) * m;
    float ang = (float)pos * inv;
    float conc = 0.1f * logf(32.f) + 1.f;
    float c = cosf(ang) * conc, s = sinf(ang) * conc;
    float x1 = qkv[off + i], x2 = qkv[off + 32 + i];
    qkv[off + i]      = x1 * c - x2 * s;
    qkv[off + 32 + i] = x2 * c + x1 * s;
}

// ---------------- Flash attention (1 wave / 16-query tile / head) ----------------
__global__ __launch_bounds__(32)
void attn_k(const float* __restrict__ qkv, const float* __restrict__ sinks,
            float* __restrict__ o, int window) {
    const int q0  = blockIdx.x * 16;
    const int h   = blockIdx.y;
    const int kvh = h >> 3;
    const int lane = threadIdx.x;
    const int half = lane >> 4, ln = lane & 15;

    __shared__ unsigned short ldsP[16 * 34];

    // Q A-fragments for head-dim steps 0..31, 32..63
    v8u qfr[2];
    const float* Qb = qkv + (q0 + ln) * QKVD + h * 64;
    #pragma unroll
    for (int s = 0; s < 2; ++s) {
        v8u a;
        #pragma unroll
        for (int v = 0; v < 8; ++v) {
            int kd = 2 * (v & 3) + 16 * (v >> 2) + 8 * half + 32 * s;
            a[v] = pack_bf(Qb[kd], Qb[kd + 1]);
        }
        qfr[s] = a;
    }

    v8f accO[4];
    #pragma unroll
    for (int nt = 0; nt < 4; ++nt) accO[nt] = zf8();
    float m_run[8], l_run[8];
    #pragma unroll
    for (int v = 0; v < 8; ++v) { m_run[v] = -3.0e38f; l_run[v] = 0.f; }

    const int jend = q0 + 15;
    int jstart = 0;
    if (window > 0) {
        int lo = q0 - window + 1; if (lo < 0) lo = 0;
        jstart = (lo >> 5) << 5;
    }
    const float* Kb = qkv + QDIM + kvh * 64;
    const float* Vb = qkv + QDIM + NKV * HDIM + kvh * 64;

    for (int j0 = jstart; j0 <= jend; j0 += 32) {
        // ---- S = Q K^T over 2 sub-tiles of 16 keys ----
        v8f sf[2];
        #pragma unroll
        for (int sub = 0; sub < 2; ++sub) {
            int key = j0 + sub * 16 + ln;
            int kc = key < (SEQC - 1) ? key : (SEQC - 1);
            const float* kp = Kb + kc * QKVD;
            v8f s = zf8();
            #pragma unroll
            for (int s2 = 0; s2 < 2; ++s2) {
                v8u b;
                #pragma unroll
                for (int v = 0; v < 8; ++v) {
                    int kd = 2 * (v & 3) + 16 * (v >> 2) + 8 * half + 32 * s2;
                    b[v] = pack_bf(kp[kd], kp[kd + 1]);
                }
                s = wmma_bf16(qfr[s2], b, s);
            }
            sf[sub] = s;
        }
        // ---- masked online softmax ----
        float rmax[8];
        #pragma unroll
        for (int v = 0; v < 8; ++v) rmax[v] = -3.0e38f;
        #pragma unroll
        for (int sub = 0; sub < 2; ++sub) {
            int key = j0 + sub * 16 + ln;
            #pragma unroll
            for (int v = 0; v < 8; ++v) {
                int qi = q0 + v + 8 * half;
                bool ok = (key <= qi) && (window <= 0 || (qi - key) < window);
                if (ok) rmax[v] = fmaxf(rmax[v], sf[sub][v] * SM_SCALE);
            }
        }
        #pragma unroll
        for (int off = 1; off < 16; off <<= 1)
            #pragma unroll
            for (int v = 0; v < 8; ++v)
                rmax[v] = fmaxf(rmax[v], __shfl_xor(rmax[v], off, 32));

        float mnew[8], sc[8], ssum[8], p[2][8];
        #pragma unroll
        for (int v = 0; v < 8; ++v) { mnew[v] = fmaxf(m_run[v], rmax[v]); ssum[v] = 0.f; }
        #pragma unroll
        for (int sub = 0; sub < 2; ++sub) {
            int key = j0 + sub * 16 + ln;
            #pragma unroll
            for (int v = 0; v < 8; ++v) {
                int qi = q0 + v + 8 * half;
                bool ok = (key <= qi) && (window <= 0 || (qi - key) < window);
                float pv = ok ? __expf(sf[sub][v] * SM_SCALE - mnew[v]) : 0.f;
                p[sub][v] = pv; ssum[v] += pv;
            }
        }
        #pragma unroll
        for (int off = 1; off < 16; off <<= 1)
            #pragma unroll
            for (int v = 0; v < 8; ++v)
                ssum[v] += __shfl_xor(ssum[v], off, 32);
        #pragma unroll
        for (int v = 0; v < 8; ++v) {
            sc[v] = __expf(m_run[v] - mnew[v]);
            l_run[v] = l_run[v] * sc[v] + ssum[v];
            m_run[v] = mnew[v];
        }
        #pragma unroll
        for (int nt = 0; nt < 4; ++nt)
            #pragma unroll
            for (int v = 0; v < 8; ++v) accO[nt][v] *= sc[v];

        // ---- transpose P (C-layout -> A-layout) through LDS ----
        __syncthreads();
        #pragma unroll
        for (int sub = 0; sub < 2; ++sub)
            #pragma unroll
            for (int v = 0; v < 8; ++v)
                ldsP[(v + 8 * half) * 34 + sub * 16 + ln] = f2bf(p[sub][v]);
        __syncthreads();
        v8u pfr;
        #pragma unroll
        for (int v = 0; v < 8; ++v) {
            int k = 2 * (v & 3) + 16 * (v >> 2) + 8 * half;
            pfr[v] = *(const unsigned int*)&ldsP[ln * 34 + k];
        }
        // ---- O += P V ----
        #pragma unroll
        for (int nt = 0; nt < 4; ++nt) {
            int d = nt * 16 + ln;
            v8u b;
            #pragma unroll
            for (int v = 0; v < 8; ++v) {
                int key = j0 + 2 * (v & 3) + 16 * (v >> 2) + 8 * half;
                int k0c = key     < SEQC ? key     : SEQC - 1;
                int k1c = key + 1 < SEQC ? key + 1 : SEQC - 1;
                b[v] = pack_bf(Vb[k0c * QKVD + d], Vb[k1c * QKVD + d]);
            }
            accO[nt] = wmma_bf16(pfr, b, accO[nt]);
        }
    }

    // ---- fold sink into denominator, store ----
    const float snk = sinks[h];
    float os[8];
    #pragma unroll
    for (int v = 0; v < 8; ++v) {
        float Mx = fmaxf(m_run[v], snk);
        float Z  = l_run[v] * __expf(m_run[v] - Mx) + __expf(snk - Mx);
        os[v] = __expf(m_run[v] - Mx) / Z;
    }
    #pragma unroll
    for (int nt = 0; nt < 4; ++nt)
        #pragma unroll
        for (int v = 0; v < 8; ++v)
            o[(q0 + v + 8 * half) * QDIM + h * 64 + nt * 16 + ln] = accO[nt][v] * os[v];
}

// ---------------- gate: logits -> top2 -> softmax route ----------------
__global__ __launch_bounds__(256)
void gate_k(const float* __restrict__ t, const float* __restrict__ gw,
            const float* __restrict__ gb, float* __restrict__ route) {
    int row = blockIdx.x;
    __shared__ float red[256][4];
    float acc[4] = {0.f, 0.f, 0.f, 0.f};
    const float* tr = t + row * DIMC;
    for (int d = threadIdx.x; d < DIMC; d += 256) {
        float tv = tr[d];
        #pragma unroll
        for (int e = 0; e < 4; ++e) acc[e] += tv * gw[d * 4 + e];
    }
    #pragma unroll
    for (int e = 0; e < 4; ++e) red[threadIdx.x][e] = acc[e];
    __syncthreads();
    for (int s = 128; s > 0; s >>= 1) {
        if (threadIdx.x < s)
            #pragma unroll
            for (int e = 0; e < 4; ++e) red[threadIdx.x][e] += red[threadIdx.x + s][e];
        __syncthreads();
    }
    if (threadIdx.x == 0) {
        float lg[4];
        #pragma unroll
        for (int e = 0; e < 4; ++e) lg[e] = red[0][e] + gb[e];
        int i1 = 0;
        for (int e = 1; e < 4; ++e) if (lg[e] > lg[i1]) i1 = e;
        int i2 = -1;
        for (int e = 0; e < 4; ++e) if (e != i1 && (i2 < 0 || lg[e] > lg[i2])) i2 = e;
        float e2 = __expf(lg[i2] - lg[i1]);
        float r[4] = {0.f, 0.f, 0.f, 0.f};
        r[i1] = 1.f / (1.f + e2);
        r[i2] = e2 / (1.f + e2);
        #pragma unroll
        for (int e = 0; e < 4; ++e) route[row * 4 + e] = r[e];
    }
}

// ---------------- clamped SiLU-GLU activation ----------------
__global__ __launch_bounds__(256)
void act_k(const float* __restrict__ h, float* __restrict__ act) {
    int idx = blockIdx.x * 256 + threadIdx.x;
    if (idx >= SEQC * INTERC) return;
    int row = idx / INTERC, i = idx - row * INTERC;
    float g = h[row * (2 * INTERC) + 2 * i];
    float l = h[row * (2 * INTERC) + 2 * i + 1];
    g = fminf(g, 7.f);
    l = fminf(fmaxf(l, -7.f), 7.f);
    float sig = 1.f / (1.f + __expf(-1.702f * g));
    act[idx] = g * sig * (l + 1.f);
}

// ---------------- host orchestration ----------------
extern "C" void kernel_launch(void* const* d_in, const int* in_sizes, int n_in,
                              void* d_out, int out_size, void* d_ws, size_t ws_size,
                              hipStream_t stream) {
    const float* x_in   = (const float*)d_in[0];
    const float* rms1   = (const float*)d_in[1];
    const float* qkv_w  = (const float*)d_in[2];
    const float* qkv_b  = (const float*)d_in[3];
    const float* out_w  = (const float*)d_in[4];
    const float* out_b  = (const float*)d_in[5];
    const float* sinks  = (const float*)d_in[6];
    const float* rms2   = (const float*)d_in[7];
    const float* gate_w = (const float*)d_in[8];
    const float* gate_b = (const float*)d_in[9];
    const float* w1     = (const float*)d_in[10];
    const float* b1     = (const float*)d_in[11];
    const float* w2     = (const float*)d_in[12];
    const float* b2     = (const float*)d_in[13];
    float* x = (float*)d_out;

    float* ws    = (float*)d_ws;
    float* t     = ws;                               // 1024*2880
    float* qkv   = t    + (size_t)SEQC * DIMC;       // 1024*5120
    float* obuf  = qkv  + (size_t)SEQC * QKVD;       // 1024*4096
    float* hbuf  = obuf + (size_t)SEQC * QDIM;       // 1024*5760
    float* abuf  = hbuf + (size_t)SEQC * 2 * INTERC; // 1024*2880
    float* route = abuf + (size_t)SEQC * INTERC;     // 1024*4

    hipMemcpyAsync(x, x_in, sizeof(float) * (size_t)SEQC * DIMC,
                   hipMemcpyDeviceToDevice, stream);

    for (int layer = 0; layer < 2; ++layer) {
        // attention block
        rmsnorm_k<<<SEQC, 256, 0, stream>>>(x, rms1 + (size_t)layer * DIMC, t);
        gemm_k<<<dim3(SEQC / BM, QKVD / BN), 256, 0, stream>>>(
            t, qkv_w + (size_t)layer * DIMC * QKVD, qkv_b + (size_t)layer * QKVD,
            nullptr, 1, qkv, SEQC, QKVD, DIMC, DIMC, QKVD, QKVD, 0);
        rope_k<<<dim3(SEQC, NHEADS + NKV), 32, 0, stream>>>(qkv);
        attn_k<<<dim3(SEQC / 16, NHEADS), 32, 0, stream>>>(
            qkv, sinks + (size_t)layer * NHEADS, obuf, (layer % 2 == 0) ? 128 : 0);
        gemm_k<<<dim3(SEQC / BM, DIMC / BN), 256, 0, stream>>>(
            obuf, out_w + (size_t)layer * QDIM * DIMC, out_b + (size_t)layer * DIMC,
            nullptr, 1, x, SEQC, DIMC, QDIM, QDIM, DIMC, DIMC, 1);

        // MoE block
        rmsnorm_k<<<SEQC, 256, 0, stream>>>(x, rms2 + (size_t)layer * DIMC, t);
        gate_k<<<SEQC, 256, 0, stream>>>(t, gate_w + (size_t)layer * DIMC * NEXP,
                                         gate_b + (size_t)layer * NEXP, route);
        for (int e = 0; e < NEXP; ++e) {
            const float* w1e = w1 + ((size_t)layer * NEXP + e) * (size_t)DIMC * 2 * INTERC;
            const float* b1e = b1 + ((size_t)layer * NEXP + e) * (size_t)2 * INTERC;
            const float* w2e = w2 + ((size_t)layer * NEXP + e) * (size_t)INTERC * DIMC;
            const float* b2e = b2 + ((size_t)layer * NEXP + e) * (size_t)DIMC;
            gemm_k<<<dim3(SEQC / BM, (2 * INTERC) / BN), 256, 0, stream>>>(
                t, w1e, b1e, nullptr, 1, hbuf,
                SEQC, 2 * INTERC, DIMC, DIMC, 2 * INTERC, 2 * INTERC, 0);
            act_k<<<(SEQC * INTERC + 255) / 256, 256, 0, stream>>>(hbuf, abuf);
            gemm_k<<<dim3(SEQC / BM, DIMC / BN), 256, 0, stream>>>(
                abuf, w2e, b2e, route + e, 4, x,
                SEQC, DIMC, INTERC, INTERC, DIMC, DIMC, 1);
        }
    }
}
